// FactualAttention_37915971289169
// MI455X (gfx1250) — matrix operand order repaired and checked by reading, not compiled, not run on hardware
//
#include <hip/hip_runtime.h>
#include <hip/hip_bf16.h>

// ---------------------------------------------------------------------------
// FactualAttention: out = A2 @ (embed^T @ Wv^T)
//   A2 lower-triangular Toeplitz: A2[i,j] = w[i-j] (i>=j) else 0
// v_wmma_f32_16x16x32_bf16 for math; TDM (tensor_load_to_lds) for tile
// staging with double buffering; A2 tiles regenerated from a 1-D table.
// ---------------------------------------------------------------------------

typedef unsigned short u16;
typedef __attribute__((ext_vector_type(16))) __bf16       v16bf;
typedef __attribute__((ext_vector_type(16))) u16          v16u;
typedef __attribute__((ext_vector_type(8)))  float        v8f;
typedef __attribute__((ext_vector_type(4)))  unsigned int u32x4;
typedef __attribute__((ext_vector_type(8)))  int          i32x8;
typedef __attribute__((ext_vector_type(4)))  int          i32x4;

#define LTOT 4096   // SERIES_LEN
#define DE   1024   // D_EMB
#define DV   1024   // D_V

constexpr int BM  = 128;
constexpr int BN  = 128;
constexpr int BK  = 32;
constexpr int LDA = BK + 8;   // [m][k] tiles: 32 elems + 4-dword pad -> 80B rows
constexpr int LDB = BN + 8;   // [k][n] tile:  128 elems + 4-dword pad -> 272B rows

__device__ __forceinline__ u16 f2bf(float x) {
    union { float f; unsigned u; } v; v.f = x;
    unsigned r = v.u + 0x7FFFu + ((v.u >> 16) & 1u);   // round-to-nearest-even
    return (u16)(r >> 16);
}

// ---- TDM: 2-D bf16 tile load, global -> LDS, with LDS row padding ---------
// pad_int: pad_interval code (3 -> every 16 dwords, 5 -> every 64 dwords)
// pad_amt: pad_amount  code (3 -> 4 dwords = 8 u16)
__device__ __forceinline__ void tdm_load_2d(unsigned lds_addr, const void* gptr,
                                            unsigned tensor_d0, unsigned tensor_d1,
                                            unsigned tile_d0,   unsigned tile_d1,
                                            unsigned stride_d0,
                                            unsigned pad_int,   unsigned pad_amt) {
    unsigned long long ga = (unsigned long long)(uintptr_t)gptr;
    u32x4 g0;
    g0[0] = 1u;                                        // count=1, user descriptor
    g0[1] = lds_addr;                                  // lds_addr  (bits 63:32)
    g0[2] = (unsigned)(ga & 0xFFFFFFFFu);              // global_addr[31:0]
    g0[3] = (unsigned)((ga >> 32) & 0x01FFFFFFu)       // global_addr[56:32]
          | (2u << 30);                                // type = 2 (image)
    i32x8 g1;
    g1[0] = (int)((1u << 16)                           // data_size = 1 -> 2 bytes
          | (1u << 20)                                 // pad_enable
          | (pad_int << 22) | (pad_amt << 25));
    g1[1] = (int)((tensor_d0 & 0xFFFFu) << 16);        // tensor_dim0[15:0]
    g1[2] = (int)(((tensor_d0 >> 16) & 0xFFFFu) | ((tensor_d1 & 0xFFFFu) << 16));
    g1[3] = (int)(((tensor_d1 >> 16) & 0xFFFFu) | ((tile_d0 & 0xFFFFu) << 16));
    g1[4] = (int)(tile_d1 & 0xFFFFu);                  // tile_dim1 ; tile_dim2 = 0
    g1[5] = (int)stride_d0;                            // tensor_dim0_stride[31:0]
    g1[6] = 0;
    g1[7] = 0;
    i32x4 z4 = {0, 0, 0, 0};                           // groups 2/3 unused (2-D tile)
    i32x8 z8 = {0, 0, 0, 0, 0, 0, 0, 0};
    __builtin_amdgcn_tensor_load_to_lds(g0, g1, z4, z4, z8, 0);
}

// --------------------------- kernel 0: w table -----------------------------
__global__ void fattn_wtab_kernel(float* __restrict__ w) {
    int d = blockIdx.x * 256 + threadIdx.x;
    if (d >= LTOT) return;
    const float Hf  = 0.2f;                  // hurst - 0.5
    const float dt  = 1.0f / (float)LTOT;
    const float eps = 1e-8f;
    float base = (d == 0) ? eps : ((float)d * dt + eps);
    w[d] = __powf(base, Hf);
}

// ---------------- prep: embed (E,L) f32 -> EbfT (L,E) bf16 -----------------
__global__ __launch_bounds__(256)
void fattn_prep_embed(const float* __restrict__ embed, u16* __restrict__ EbfT) {
    __shared__ u16 tile[64][68];
    const int t  = threadIdx.x;
    const int e0 = blockIdx.x * 64;
    const int l0 = blockIdx.y * 64;
    #pragma unroll
    for (int p = 0; p < 16; ++p) {
        int idx = t + p * 256;          // 0..4095
        int ee  = idx >> 6;
        int ll  = idx & 63;             // contiguous in memory -> coalesced read
        tile[ll][ee] = f2bf(embed[(size_t)(e0 + ee) * LTOT + l0 + ll]);
    }
    __syncthreads();
    #pragma unroll
    for (int p = 0; p < 16; ++p) {
        int idx = t + p * 256;
        int ll  = idx >> 6;
        int ee  = idx & 63;             // contiguous -> coalesced write
        EbfT[(size_t)(l0 + ll) * DE + e0 + ee] = tile[ll][ee];
    }
}

// ---------------- prep: Wv (V,E) f32 -> bf16 (same layout) -----------------
__global__ __launch_bounds__(256)
void fattn_prep_wv(const float* __restrict__ Wv, u16* __restrict__ Wvbf) {
    size_t i = (size_t)blockIdx.x * 256 + threadIdx.x;   // one float4 each
    const float4 v = *(const float4*)(Wv + i * 4);
    uint2 r;
    r.x = (unsigned)f2bf(v.x) | ((unsigned)f2bf(v.y) << 16);
    r.y = (unsigned)f2bf(v.z) | ((unsigned)f2bf(v.w) << 16);
    *(uint2*)(Wvbf + i * 4) = r;
}

// --------------------------- kernel 1: Y = E^T @ W^T -----------------------
// EbfT: (L,E) bf16 ; Wvbf: (V,E) bf16 ; Ybf: (L,V) bf16
__global__ __launch_bounds__(256)
void fattn_gemm1_kernel(const u16* __restrict__ EbfT,
                        const u16* __restrict__ Wvbf,
                        u16* __restrict__ Ybf) {
    __shared__ __align__(16) u16 As[2][BM][LDA];   // [l][e]
    __shared__ __align__(16) u16 Bs[2][BN][LDA];   // [v][e]

    const int t    = threadIdx.x;
    const int lane = t & 31;
    const int half = lane >> 4;
    const int l15  = lane & 15;
    const int wave = t >> 5;
    const int wm   = wave >> 2;
    const int wn   = wave & 3;
    const int l0   = blockIdx.x * BM;
    const int n0   = blockIdx.y * BN;
    const int wsg  = __builtin_amdgcn_readfirstlane(t) >> 5;  // SGPR wave id

    v8f acc[4][2] = {};

    if (wsg == 0) {   // scalar branch: only wave 0 drives the TDM
        tdm_load_2d((unsigned)(uintptr_t)&As[0][0][0], EbfT + (size_t)l0 * DE,
                    DE, LTOT, BK, BM, DE, 3, 3);
        tdm_load_2d((unsigned)(uintptr_t)&Bs[0][0][0], Wvbf + (size_t)n0 * DE,
                    DE, DV, BK, BM, DE, 3, 3);
    }

    const int nsteps = DE / BK;
    for (int ks = 0; ks < nsteps; ++ks) {
        __syncthreads();   // compute(ks-1) finished before TDM reuses its buffer
        if (wsg == 0) {
            if (ks + 1 < nsteps) {
                const int nk = (ks + 1) * BK;
                tdm_load_2d((unsigned)(uintptr_t)&As[(ks + 1) & 1][0][0],
                            EbfT + (size_t)l0 * DE + nk, DE, LTOT, BK, BM, DE, 3, 3);
                tdm_load_2d((unsigned)(uintptr_t)&Bs[(ks + 1) & 1][0][0],
                            Wvbf + (size_t)n0 * DE + nk, DE, DV, BK, BM, DE, 3, 3);
                __builtin_amdgcn_s_wait_tensorcnt((short)2);  // oldest 2 (step ks) done
            } else {
                __builtin_amdgcn_s_wait_tensorcnt((short)0);
            }
        }
        __syncthreads();   // tile ks visible to all waves
        const u16 (*Ab)[LDA] = As[ks & 1];
        const u16 (*Bb)[LDA] = Bs[ks & 1];

        v16bf a[4], b[2];
        #pragma unroll
        for (int mt = 0; mt < 4; ++mt) {
            const u16* pa = &Ab[wm * 64 + mt * 16 + l15][half * 8];
            v16u tmp;
            #pragma unroll
            for (int i = 0; i < 8; ++i) { tmp[i] = pa[i]; tmp[8 + i] = pa[16 + i]; }
            a[mt] = __builtin_bit_cast(v16bf, tmp);
        }
        #pragma unroll
        for (int nt = 0; nt < 2; ++nt) {
            const u16* pb = &Bb[wn * 32 + nt * 16 + l15][half * 8];
            v16u tmp;
            #pragma unroll
            for (int i = 0; i < 8; ++i) { tmp[i] = pb[i]; tmp[8 + i] = pb[16 + i]; }
            b[nt] = __builtin_bit_cast(v16bf, tmp);
        }
        #pragma unroll
        for (int mt = 0; mt < 4; ++mt)
            #pragma unroll
            for (int nt = 0; nt < 2; ++nt)
                acc[mt][nt] = __builtin_amdgcn_wmma_f32_16x16x32_bf16(
                    false, a[mt], false, b[nt], (short)0, acc[mt][nt], false, false);
    }

    #pragma unroll
    for (int mt = 0; mt < 4; ++mt) {
        int row = l0 + wm * 64 + mt * 16 + half * 8;
        #pragma unroll
        for (int nt = 0; nt < 2; ++nt) {
            int col = n0 + wn * 32 + nt * 16 + l15;
            #pragma unroll
            for (int r = 0; r < 8; ++r)
                Ybf[(size_t)(row + r) * DV + col] = f2bf(acc[mt][nt][r]);
        }
    }
}

// --------------------------- kernel 2: out = A2 @ Y ------------------------
__global__ __launch_bounds__(256)
void fattn_gemm2_kernel(const float* __restrict__ wtab,
                        const u16* __restrict__ Ybf,
                        float* __restrict__ out) {
    __shared__ __align__(16) u16 As[BM][LDA];      // A2 tile (regenerated)
    __shared__ __align__(16) u16 Bs[2][BK][LDB];   // Y tile via TDM

    const int t    = threadIdx.x;
    const int lane = t & 31;
    const int half = lane >> 4;
    const int l15  = lane & 15;
    const int wave = t >> 5;
    const int wm   = wave >> 2;
    const int wn   = wave & 3;
    const int bm   = (int)gridDim.x - 1 - (int)blockIdx.x;   // long K rows first
    const int i0   = bm * BM;
    const int n0   = blockIdx.y * BN;
    const int wsg  = __builtin_amdgcn_readfirstlane(t) >> 5;
    const int nsteps = (i0 + BM) / BK;             // causal: j <= i only

    v8f acc[4][2] = {};

    if (wsg == 0) {
        tdm_load_2d((unsigned)(uintptr_t)&Bs[0][0][0], Ybf + n0,
                    DV, LTOT, BN, BK, DV, 5, 3);
    }

    for (int ks = 0; ks < nsteps; ++ks) {
        __syncthreads();
        const int j0 = ks * BK;
        // regenerate A2 tile from the 1-D Toeplitz table (overlaps with TDM)
        for (int e = t; e < BM * BK; e += 256) {
            int ii = e >> 5;
            int jj = e & 31;
            int d  = (i0 + ii) - (j0 + jj);
            float w = (d < 0) ? 0.0f : wtab[d];
            As[ii][jj] = f2bf(w);
        }
        if (wsg == 0) {
            if (ks + 1 < nsteps) {
                tdm_load_2d((unsigned)(uintptr_t)&Bs[(ks + 1) & 1][0][0],
                            Ybf + (size_t)(ks + 1) * BK * DV + n0,
                            DV, LTOT, BN, BK, DV, 5, 3);
                __builtin_amdgcn_s_wait_tensorcnt((short)1);
            } else {
                __builtin_amdgcn_s_wait_tensorcnt((short)0);
            }
        }
        __syncthreads();
        const u16 (*Bb)[LDB] = Bs[ks & 1];

        v16bf a[4], b[2];
        #pragma unroll
        for (int mt = 0; mt < 4; ++mt) {
            const u16* pa = &As[wm * 64 + mt * 16 + l15][half * 8];
            v16u tmp;
            #pragma unroll
            for (int i = 0; i < 8; ++i) { tmp[i] = pa[i]; tmp[8 + i] = pa[16 + i]; }
            a[mt] = __builtin_bit_cast(v16bf, tmp);
        }
        #pragma unroll
        for (int nt = 0; nt < 2; ++nt) {
            int col = wn * 32 + nt * 16 + l15;
            v16u tmp;
            #pragma unroll
            for (int i = 0; i < 8; ++i) { tmp[i] = Bb[half * 8 + i][col]; tmp[8 + i] = Bb[16 + half * 8 + i][col]; }
            b[nt] = __builtin_bit_cast(v16bf, tmp);
        }
        #pragma unroll
        for (int mt = 0; mt < 4; ++mt)
            #pragma unroll
            for (int nt = 0; nt < 2; ++nt)
                acc[mt][nt] = __builtin_amdgcn_wmma_f32_16x16x32_bf16(
                    false, a[mt], false, b[nt], (short)0, acc[mt][nt], false, false);
    }

    #pragma unroll
    for (int mt = 0; mt < 4; ++mt) {
        int row = i0 + wm * 64 + mt * 16 + half * 8;
        #pragma unroll
        for (int nt = 0; nt < 2; ++nt) {
            int col = n0 + wn * 32 + nt * 16 + l15;
            #pragma unroll
            for (int r = 0; r < 8; ++r)
                out[(size_t)(row + r) * DV + col] = acc[mt][nt][r];
        }
    }
}

// ---------------------------------------------------------------------------
extern "C" void kernel_launch(void* const* d_in, const int* in_sizes, int n_in,
                              void* d_out, int out_size, void* d_ws, size_t ws_size,
                              hipStream_t stream) {
    const float* embed = (const float*)d_in[0];   // (1024, 4096) f32
    const float* Wv    = (const float*)d_in[1];   // (1024, 1024) f32
    float*       out   = (float*)d_out;           // (4096, 1024) f32

    // workspace: [0,16K) wtab | [64K, +8M) EbfT | [+8M, +10M) Wvbf | [+10M, +18M) Ybf
    char* ws = (char*)d_ws;
    float* wtab = (float*)ws;
    u16*   EbfT = (u16*)(ws + (64 << 10));
    u16*   Wvbf = (u16*)(ws + (64 << 10) + (8u << 20));
    u16*   Ybf  = (u16*)(ws + (64 << 10) + (10u << 20));

    fattn_wtab_kernel <<<LTOT / 256, 256, 0, stream>>>(wtab);
    fattn_prep_embed  <<<dim3(DE / 64, LTOT / 64), 256, 0, stream>>>(embed, EbfT);
    fattn_prep_wv     <<<(DV * DE / 4) / 256, 256, 0, stream>>>(Wv, Wvbf);
    fattn_gemm1_kernel<<<dim3(LTOT / BM, DV / BN), 256, 0, stream>>>(EbfT, Wvbf, Ybf);
    fattn_gemm2_kernel<<<dim3(LTOT / BM, DV / BN), 256, 0, stream>>>(wtab, Ybf, out);
}
